// PatchedDbrxExperts_33251636805988
// MI455X (gfx1250) — matrix operand order repaired and checked by reading, not compiled, not run on hardware
//
#include <hip/hip_runtime.h>
#include <stdint.h>

// ---------------- problem dims (DBRX MoE FFN) ----------------
#define NTOK 2048   // B*S tokens
#define HID  2048   // hidden size H
#define FFN  4096   // ffn hidden F
#define NEXP 8
#define TOPK 4

// ---------------- CDNA5 feature detection ----------------
#ifndef __has_builtin
#define __has_builtin(x) 0
#endif

#if __has_builtin(__builtin_amdgcn_tensor_load_to_lds)
#define HAVE_TDM 1
#else
#define HAVE_TDM 0
#endif

#if defined(__has_include)
#if __has_include(<hip/amd_detail/amd_gfx1250_TDM.h>)
#define TDM_SIX_ARG 1
#endif
#endif
#ifndef TDM_SIX_ARG
#define TDM_SIX_ARG 0
#endif

#if __has_builtin(__builtin_amdgcn_s_wait_tensorcnt)
#define HAVE_WAIT_TCNT 1
#else
#define HAVE_WAIT_TCNT 0
#endif

// ---------------- types ----------------
typedef __bf16 v16bf __attribute__((ext_vector_type(16)));
typedef float  v8f   __attribute__((ext_vector_type(8)));

union FragBF { uint4 u4[2]; v16bf v; };

// LDS tile row stride: 32 bf16 (64B) + 16B pad = 80B (16B aligned, conflict-free)
#define LDS_STRIDE 80

// ---------------- scalar helpers ----------------
#if __has_builtin(__builtin_amdgcn_cvt_pk_bf16_f32)
typedef __bf16 v2bf __attribute__((ext_vector_type(2)));
__device__ __forceinline__ uint32_t pack2_bf16(float lo, float hi) {
  union { v2bf v; uint32_t u; } c;
  c.v = __builtin_amdgcn_cvt_pk_bf16_f32(lo, hi);
  return c.u;
}
__device__ __forceinline__ uint32_t f32_to_bf16_bits(float f) {
  return pack2_bf16(f, 0.f) & 0xFFFFu;
}
#else
// round-half-up + v_perm_b32: 3 VALU per pair (vs ~12 for the RNE bit dance)
__device__ __forceinline__ uint32_t pack2_bf16(float lo, float hi) {
  uint32_t ulo = __float_as_uint(lo) + 0x8000u;
  uint32_t uhi = __float_as_uint(hi) + 0x8000u;
  return __builtin_amdgcn_perm(uhi, ulo, 0x07060302u);  // {uhi[31:16], ulo[31:16]}
}
__device__ __forceinline__ uint32_t f32_to_bf16_bits(float f) {
  return (__float_as_uint(f) + 0x8000u) >> 16;          // round half up
}
#endif

// silu(g)*u with fast v_rcp_f32 (no IEEE divide expansion)
__device__ __forceinline__ float silu_mul(float g, float u) {
  float s = __builtin_amdgcn_rcpf(1.0f + __expf(-g));
  return g * s * u;
}

// ---------------- WMMA fragment helpers ----------------
// A fragment: 16x32 bf16. Lanes 0-15 row M=lane hold K 0..7 & 16..23,
// lanes 16-31 row M=lane-16 hold K 8..15 & 24..31 (ISA 16-bit A layout).
__device__ __forceinline__ v16bf load_a_frag(const uint8_t* lds, int row0, int lane) {
  int r  = row0 + (lane & 15);
  int kb = (lane >> 4) * 16;
  FragBF f;
  f.u4[0] = *(const uint4*)(lds + r * LDS_STRIDE + kb);
  f.u4[1] = *(const uint4*)(lds + r * LDS_STRIDE + 32 + kb);
  return f.v;
}
// B fragment: 32x16 bf16. Lanes 0-15 col N=lane hold K 0..7 & 16..23, lanes 16-31 K 8..15 & 24..31.
__device__ __forceinline__ v16bf load_b_frag(const uint8_t* lds, int row0, int lane) {
  int r  = row0 + (lane & 15);
  int kb = (lane >> 4) * 16;
  FragBF f;
  f.u4[0] = *(const uint4*)(lds + r * LDS_STRIDE + kb);
  f.u4[1] = *(const uint4*)(lds + r * LDS_STRIDE + 32 + kb);
  return f.v;
}

__device__ __forceinline__ v8f wmma_bf16(v16bf a, v16bf b, v8f c) {
  return __builtin_amdgcn_wmma_f32_16x16x32_bf16(
      false, a, false, b, (short)0, c, false, false);
}

// ---------------- TDM (Tensor Data Mover) ----------------
#if HAVE_TDM
typedef unsigned int u32x4 __attribute__((ext_vector_type(4)));
typedef int          i32x4 __attribute__((ext_vector_type(4)));
typedef int          i32x8 __attribute__((ext_vector_type(8)));

// Load a [tileRows x tileK] bf16 tile (global row stride = strideElems) into LDS,
// padding +16B after every 64B row -> LDS_STRIDE layout. D# per ISA ch.8.
__device__ __forceinline__ void tdm_load_tile_bf16(uint32_t lds_byte_addr,
                                                   const void* gsrc,
                                                   uint32_t tileK, uint32_t tileRows,
                                                   uint32_t strideElems,
                                                   uint32_t tensorRows) {
  unsigned long long ga = (unsigned long long)(uintptr_t)gsrc;
  u32x4 g0;
  g0[0] = 1u;                                                // count=1, user mode
  g0[1] = lds_byte_addr;                                     // lds_addr
  g0[2] = (uint32_t)ga;                                      // global_addr[31:0]
  g0[3] = (uint32_t)((ga >> 32) & 0x01FFFFFFu) | (2u << 30); // addr[56:32] | type=2
  i32x8 g1;
  // data_size=2B | pad_enable | pad_interval=3 (16 dw = 64B) | pad_amount=3 (4 dw = 16B)
  g1[0] = (int)((1u << 16) | (1u << 20) | (3u << 22) | (3u << 25));
  g1[1] = (int)((strideElems & 0xFFFFu) << 16);                        // tensor_dim0 lo
  g1[2] = (int)((strideElems >> 16) | ((tensorRows & 0xFFFFu) << 16)); // dim0 hi | dim1 lo
  g1[3] = (int)((tensorRows >> 16) | (tileK << 16));                   // dim1 hi | tile_dim0
  g1[4] = (int)(tileRows & 0xFFFFu);                                   // tile_dim1
  g1[5] = (int)strideElems;                                            // dim0_stride lo
  g1[6] = 0;
  g1[7] = 0;
  i32x4 z4 = {0, 0, 0, 0};
#if TDM_SIX_ARG
  i32x8 z8 = {0, 0, 0, 0, 0, 0, 0, 0};
  __builtin_amdgcn_tensor_load_to_lds(g0, g1, z4, z4, z8, 0);
#else
  __builtin_amdgcn_tensor_load_to_lds(g0, g1, z4, z4, 0);
#endif
}

__device__ __forceinline__ void wait_tdm() {
#if HAVE_WAIT_TCNT
  __builtin_amdgcn_s_wait_tensorcnt(0);
#else
  asm volatile("s_wait_tensorcnt 0x0" ::: "memory");
#endif
}
#endif  // HAVE_TDM

// stage a 64x32 weight tile row (one thread: 8 f32 -> 8 bf16) into LDS
__device__ __forceinline__ void stage_w_row(uint8_t* dst, float4 a, float4 b,
                                            int rr, int cc) {
  *(uint2*)(dst + rr * LDS_STRIDE + cc * 2) =
      make_uint2(pack2_bf16(a.x, a.y), pack2_bf16(a.z, a.w));
  *(uint2*)(dst + rr * LDS_STRIDE + cc * 2 + 8) =
      make_uint2(pack2_bf16(b.x, b.y), pack2_bf16(b.z, b.w));
}

// transpose-stage 8 h-values of one w2 k-column into LDS [64 h][32 k]
__device__ __forceinline__ void stage_w2t(uint8_t* dst, float4 a, float4 b,
                                          int f, int hc) {
  *(uint16_t*)(dst + (hc + 0) * LDS_STRIDE + f * 2) = (uint16_t)f32_to_bf16_bits(a.x);
  *(uint16_t*)(dst + (hc + 1) * LDS_STRIDE + f * 2) = (uint16_t)f32_to_bf16_bits(a.y);
  *(uint16_t*)(dst + (hc + 2) * LDS_STRIDE + f * 2) = (uint16_t)f32_to_bf16_bits(a.z);
  *(uint16_t*)(dst + (hc + 3) * LDS_STRIDE + f * 2) = (uint16_t)f32_to_bf16_bits(a.w);
  *(uint16_t*)(dst + (hc + 4) * LDS_STRIDE + f * 2) = (uint16_t)f32_to_bf16_bits(b.x);
  *(uint16_t*)(dst + (hc + 5) * LDS_STRIDE + f * 2) = (uint16_t)f32_to_bf16_bits(b.y);
  *(uint16_t*)(dst + (hc + 6) * LDS_STRIDE + f * 2) = (uint16_t)f32_to_bf16_bits(b.z);
  *(uint16_t*)(dst + (hc + 7) * LDS_STRIDE + f * 2) = (uint16_t)f32_to_bf16_bits(b.w);
}

// ---------------- prep kernels ----------------
__global__ void k_prep_xb(const float* __restrict__ x, uint32_t* __restrict__ xb, int n2) {
  int i = blockIdx.x * blockDim.x + threadIdx.x;
  if (i < n2) {
    float2 v = ((const float2*)x)[i];
    xb[i] = pack2_bf16(v.x, v.y);
  }
}

__global__ void k_prep_scale(const float* __restrict__ tw, const int* __restrict__ te,
                             float* __restrict__ scale) {
  int t = blockIdx.x * blockDim.x + threadIdx.x;
  if (t < NTOK) {
    float s[NEXP];
#pragma unroll
    for (int e = 0; e < NEXP; ++e) s[e] = 0.f;
#pragma unroll
    for (int k = 0; k < TOPK; ++k) {
      int   e = te[t * TOPK + k];
      float w = tw[t * TOPK + k];
#pragma unroll
      for (int j = 0; j < NEXP; ++j)
        if (j == e) s[j] += w;
    }
#pragma unroll
    for (int e = 0; e < NEXP; ++e) scale[t * NEXP + e] = s[e];
  }
}

// ---------------- stage 1: h = silu(x w1^T) * (x v1^T) ----------------
// grid: (FFN/64, NTOK/128); block 256 = 8 wave32 (4 M-waves x 2 N-waves, 32x32/wave)
// Software-pipelined: TDM/next-tile VMEM loads issued before this tile's WMMAs,
// double-buffered LDS, one barrier per K step.
__global__ void __launch_bounds__(256)
k_gateup(const uint16_t* __restrict__ xb, const float* __restrict__ w1e,
         const float* __restrict__ v1e, uint16_t* __restrict__ hb) {
  __shared__ uint8_t sA[2][128 * LDS_STRIDE];
  __shared__ uint8_t sW[2][64 * LDS_STRIDE];
  __shared__ uint8_t sV[2][64 * LDS_STRIDE];

  const int tid = threadIdx.x, lane = tid & 31, wid = tid >> 5;
  const int wm = wid & 3, wn = wid >> 2;
  const int m0 = blockIdx.y * 128;   // token tile base
  const int f0 = blockIdx.x * 64;    // ffn tile base
  const int rr = tid >> 2, cc = (tid & 3) * 8;   // weight staging coords

  v8f accG[2][2] = {}, accU[2][2] = {};

  // ---- prologue: stage k0 = 0 into buffer 0 ----
#if HAVE_TDM
  tdm_load_tile_bf16((uint32_t)(uintptr_t)&sA[0][0] + (uint32_t)(wid * 16 * LDS_STRIDE),
                     xb + (size_t)(m0 + wid * 16) * HID, 32u, 16u,
                     (uint32_t)HID, (uint32_t)NTOK);
#else
#pragma unroll
  for (int j = 0; j < 2; ++j) {
    int id = tid + j * 256;
    int r = id >> 2, c = id & 3;
    uint4 v = *(const uint4*)(xb + (size_t)(m0 + r) * HID + c * 8);
    *(uint4*)(&sA[0][0] + r * LDS_STRIDE + c * 16) = v;
  }
#endif
  {
    const float* g0p = w1e + (size_t)(f0 + rr) * HID + cc;
    stage_w_row(&sW[0][0], *(const float4*)g0p, *(const float4*)(g0p + 4), rr, cc);
    const float* g1p = v1e + (size_t)(f0 + rr) * HID + cc;
    stage_w_row(&sV[0][0], *(const float4*)g1p, *(const float4*)(g1p + 4), rr, cc);
  }

#pragma unroll 2
  for (int k0 = 0; k0 < HID; k0 += 32) {
    const int  b        = (k0 >> 5) & 1;
    const bool has_next = (k0 + 32) < HID;

#if HAVE_TDM
    wait_tdm();             // this buffer's tile DMA complete (own slice)
#endif
    __syncthreads();        // all slices + weight stores visible; old reads retired

    // ---- issue next tile: TDM + raw weight loads (latency hidden under WMMA) ----
    float4 wA0 = {}, wA1 = {}, wB0 = {}, wB1 = {};
#if !HAVE_TDM
    uint4 xn0 = {}, xn1 = {};
#endif
    if (has_next) {
      const int kn = k0 + 32;
#if HAVE_TDM
      tdm_load_tile_bf16((uint32_t)(uintptr_t)&sA[b ^ 1][0] +
                             (uint32_t)(wid * 16 * LDS_STRIDE),
                         xb + (size_t)(m0 + wid * 16) * HID + kn, 32u, 16u,
                         (uint32_t)HID, (uint32_t)NTOK);
#else
      {
        int id0 = tid, id1 = tid + 256;
        xn0 = *(const uint4*)(xb + (size_t)(m0 + (id0 >> 2)) * HID + kn + (id0 & 3) * 8);
        xn1 = *(const uint4*)(xb + (size_t)(m0 + (id1 >> 2)) * HID + kn + (id1 & 3) * 8);
      }
#endif
      const float* g0p = w1e + (size_t)(f0 + rr) * HID + kn + cc;
      wA0 = *(const float4*)g0p;
      wA1 = *(const float4*)(g0p + 4);
      const float* g1p = v1e + (size_t)(f0 + rr) * HID + kn + cc;
      wB0 = *(const float4*)g1p;
      wB1 = *(const float4*)(g1p + 4);
    }

    // ---- WMMA on buffer b ----
    {
      const uint8_t* A = &sA[b][0];
      const uint8_t* W = &sW[b][0];
      const uint8_t* V = &sV[b][0];
      v16bf a0  = load_a_frag(A, wm * 32 + 0,  lane);
      v16bf a1  = load_a_frag(A, wm * 32 + 16, lane);
      v16bf bw0 = load_b_frag(W, wn * 32 + 0,  lane);
      v16bf bw1 = load_b_frag(W, wn * 32 + 16, lane);
      v16bf bv0 = load_b_frag(V, wn * 32 + 0,  lane);
      v16bf bv1 = load_b_frag(V, wn * 32 + 16, lane);

      accG[0][0] = wmma_bf16(a0, bw0, accG[0][0]);
      accG[0][1] = wmma_bf16(a0, bw1, accG[0][1]);
      accG[1][0] = wmma_bf16(a1, bw0, accG[1][0]);
      accG[1][1] = wmma_bf16(a1, bw1, accG[1][1]);
      accU[0][0] = wmma_bf16(a0, bv0, accU[0][0]);
      accU[0][1] = wmma_bf16(a0, bv1, accU[0][1]);
      accU[1][0] = wmma_bf16(a1, bv0, accU[1][0]);
      accU[1][1] = wmma_bf16(a1, bv1, accU[1][1]);
    }

    // ---- convert + stage next tiles into back buffer ----
    if (has_next) {
      stage_w_row(&sW[b ^ 1][0], wA0, wA1, rr, cc);
      stage_w_row(&sV[b ^ 1][0], wB0, wB1, rr, cc);
#if !HAVE_TDM
      {
        int id0 = tid, id1 = tid + 256;
        *(uint4*)(&sA[b ^ 1][0] + (id0 >> 2) * LDS_STRIDE + (id0 & 3) * 16) = xn0;
        *(uint4*)(&sA[b ^ 1][0] + (id1 >> 2) * LDS_STRIDE + (id1 & 3) * 16) = xn1;
      }
#endif
    }
  }

  // ---- epilogue: silu(g)*u -> bf16 hb[t, f] ----
  const int n_ = lane & 15, mh = (lane >> 4) * 8;
#pragma unroll
  for (int i = 0; i < 2; ++i) {
    int tb = m0 + wm * 32 + i * 16 + mh;
#pragma unroll
    for (int j = 0; j < 2; ++j) {
      int f = f0 + wn * 32 + j * 16 + n_;
#pragma unroll
      for (int r = 0; r < 8; ++r) {
        float h = silu_mul(accG[i][j][r], accU[i][j][r]);
        hb[(size_t)(tb + r) * FFN + f] = (uint16_t)f32_to_bf16_bits(h);
      }
    }
  }
}

// ---------------- stage 2: out (+)= (hb @ w2) * scale[:,e] ----------------
// grid: (HID/64, NTOK/128); same pipelined structure, w2 transposed through LDS.
__global__ void __launch_bounds__(256)
k_down(const uint16_t* __restrict__ hb, const float* __restrict__ w2e,
       const float* __restrict__ scale, int expert, int beta,
       float* __restrict__ out) {
  __shared__ uint8_t sA[2][128 * LDS_STRIDE];
  __shared__ uint8_t sB[2][64 * LDS_STRIDE];

  const int tid = threadIdx.x, lane = tid & 31, wid = tid >> 5;
  const int wm = wid & 3, wn = wid >> 2;
  const int m0 = blockIdx.y * 128;   // token tile base
  const int h0 = blockIdx.x * 64;    // hidden tile base
  const int fc = tid >> 3, hc = (tid & 7) * 8;  // w2 transpose-staging coords

  v8f acc[2][2] = {};

  // ---- prologue: stage k0 = 0 into buffer 0 ----
#if HAVE_TDM
  tdm_load_tile_bf16((uint32_t)(uintptr_t)&sA[0][0] + (uint32_t)(wid * 16 * LDS_STRIDE),
                     hb + (size_t)(m0 + wid * 16) * FFN, 32u, 16u,
                     (uint32_t)FFN, (uint32_t)NTOK);
#else
#pragma unroll
  for (int j = 0; j < 2; ++j) {
    int id = tid + j * 256;
    int r = id >> 2, c = id & 3;
    uint4 v = *(const uint4*)(hb + (size_t)(m0 + r) * FFN + c * 8);
    *(uint4*)(&sA[0][0] + r * LDS_STRIDE + c * 16) = v;
  }
#endif
  {
    const float* gp = w2e + (size_t)fc * HID + h0 + hc;
    stage_w2t(&sB[0][0], *(const float4*)gp, *(const float4*)(gp + 4), fc, hc);
  }

#pragma unroll 2
  for (int k0 = 0; k0 < FFN; k0 += 32) {
    const int  b        = (k0 >> 5) & 1;
    const bool has_next = (k0 + 32) < FFN;

#if HAVE_TDM
    wait_tdm();
#endif
    __syncthreads();

    float4 wA = {}, wB = {};
#if !HAVE_TDM
    uint4 xn0 = {}, xn1 = {};
#endif
    if (has_next) {
      const int kn = k0 + 32;
#if HAVE_TDM
      tdm_load_tile_bf16((uint32_t)(uintptr_t)&sA[b ^ 1][0] +
                             (uint32_t)(wid * 16 * LDS_STRIDE),
                         hb + (size_t)(m0 + wid * 16) * FFN + kn, 32u, 16u,
                         (uint32_t)FFN, (uint32_t)NTOK);
#else
      {
        int id0 = tid, id1 = tid + 256;
        xn0 = *(const uint4*)(hb + (size_t)(m0 + (id0 >> 2)) * FFN + kn + (id0 & 3) * 8);
        xn1 = *(const uint4*)(hb + (size_t)(m0 + (id1 >> 2)) * FFN + kn + (id1 & 3) * 8);
      }
#endif
      const float* gp = w2e + (size_t)(kn + fc) * HID + h0 + hc;
      wA = *(const float4*)gp;
      wB = *(const float4*)(gp + 4);
    }

    {
      const uint8_t* A  = &sA[b][0];
      const uint8_t* Bt = &sB[b][0];
      v16bf a0 = load_a_frag(A, wm * 32 + 0,  lane);
      v16bf a1 = load_a_frag(A, wm * 32 + 16, lane);
      v16bf b0 = load_b_frag(Bt, wn * 32 + 0,  lane);
      v16bf b1 = load_b_frag(Bt, wn * 32 + 16, lane);

      acc[0][0] = wmma_bf16(a0, b0, acc[0][0]);
      acc[0][1] = wmma_bf16(a0, b1, acc[0][1]);
      acc[1][0] = wmma_bf16(a1, b0, acc[1][0]);
      acc[1][1] = wmma_bf16(a1, b1, acc[1][1]);
    }

    if (has_next) {
      stage_w2t(&sB[b ^ 1][0], wA, wB, fc, hc);
#if !HAVE_TDM
      {
        int id0 = tid, id1 = tid + 256;
        *(uint4*)(&sA[b ^ 1][0] + (id0 >> 2) * LDS_STRIDE + (id0 & 3) * 16) = xn0;
        *(uint4*)(&sA[b ^ 1][0] + (id1 >> 2) * LDS_STRIDE + (id1 & 3) * 16) = xn1;
      }
#endif
    }
  }

  // ---- epilogue: scale & accumulate into out ----
  const int n_ = lane & 15, mh = (lane >> 4) * 8;
#pragma unroll
  for (int i = 0; i < 2; ++i) {
    int tb = m0 + wm * 32 + i * 16 + mh;
    float sc[8];
#pragma unroll
    for (int r = 0; r < 8; ++r) sc[r] = scale[(size_t)(tb + r) * NEXP + expert];
#pragma unroll
    for (int j = 0; j < 2; ++j) {
      int h = h0 + wn * 32 + j * 16 + n_;
#pragma unroll
      for (int r = 0; r < 8; ++r) {
        float v = acc[i][j][r] * sc[r];
        size_t idx = (size_t)(tb + r) * HID + h;
        out[idx] = beta ? (out[idx] + v) : v;
      }
    }
  }
}

// ---------------- host entry ----------------
extern "C" void kernel_launch(void* const* d_in, const int* in_sizes, int n_in,
                              void* d_out, int out_size, void* d_ws, size_t ws_size,
                              hipStream_t stream) {
  (void)in_sizes; (void)n_in; (void)out_size; (void)ws_size;
  const float* x     = (const float*)d_in[0];
  // d_in[1] = router probs (unused by the reference forward)
  const float* top_w = (const float*)d_in[2];
  const int*   top_e = (const int*)d_in[3];
  const float* w1    = (const float*)d_in[4];
  const float* v1    = (const float*)d_in[5];
  const float* w2    = (const float*)d_in[6];
  float* out = (float*)d_out;

  // workspace layout (~24.1 MB): xb bf16 [T,H] | scale f32 [T,E] | hb bf16 [T,F]
  uint8_t*  ws    = (uint8_t*)d_ws;
  uint32_t* xb32  = (uint32_t*)ws;
  float*    scale = (float*)(ws + (size_t)NTOK * HID * 2);
  uint16_t* hb    = (uint16_t*)(ws + (size_t)NTOK * HID * 2 + (size_t)NTOK * NEXP * 4);
  const uint16_t* xb = (const uint16_t*)xb32;

  {
    int n2 = NTOK * HID / 2;
    k_prep_xb<<<(n2 + 255) / 256, 256, 0, stream>>>(x, xb32, n2);
    k_prep_scale<<<(NTOK + 255) / 256, 256, 0, stream>>>(top_w, top_e, scale);
  }

  dim3 g1(FFN / 64, NTOK / 128);
  dim3 g2(HID / 64, NTOK / 128);
  for (int e = 0; e < NEXP; ++e) {
    k_gateup<<<g1, 256, 0, stream>>>(xb, w1 + (size_t)e * FFN * HID,
                                     v1 + (size_t)e * FFN * HID, hb);
    k_down<<<g2, 256, 0, stream>>>(hb, w2 + (size_t)e * FFN * HID,
                                   scale, e, e > 0 ? 1 : 0, out);
  }
}
